// SumConv_32100585570382
// MI455X (gfx1250) — matrix operand order repaired and checked by reading, not compiled, not run on hardware
//
#include <hip/hip_runtime.h>

typedef __attribute__((ext_vector_type(16))) _Float16 v16h;
typedef __attribute__((ext_vector_type(8)))  float    v8f;

#define N_  32
#define CI_ 32
#define CO_ 32
#define H_  64
#define W_  64
#define HW_ (H_ * W_)

// ---------------------------------------------------------------------------
// Prep: W[co,ci,kidx] = softmax_ci(logits[co,ci,k1,k2]) -> f16, scattered into
// a contiguous-K 16-bit A-matrix layout (16x32 MxK) so each lane of the main kernel
// reads its 16 halfs as one contiguous 32B chunk.
//   A-layout (contiguous K): lane g0 (0-15): halfs 0..15 -> K=0..15
//                            lane g1 (16-31): halfs 0..15 -> K=16..31
// Storage: wsA[(tile*32 + lane)*16 + half], tile = kidx*2 + (co>>4). 8 KB total.
// ---------------------------------------------------------------------------
__global__ void sumconv_prep_kernel(const float* __restrict__ logits,
                                    _Float16* __restrict__ wsA) {
    const int t    = threadIdx.x;        // 0..127
    const int co   = t >> 2;
    const int kidx = t & 3;
    const int k1   = kidx >> 1, k2 = kidx & 1;

    float l[CI_];
    float m = -INFINITY;
    #pragma unroll
    for (int ci = 0; ci < CI_; ++ci) {
        l[ci] = logits[((co * CI_ + ci) * 2 + k1) * 2 + k2];
        m = fmaxf(m, l[ci]);
    }
    float s = 0.f;
    #pragma unroll
    for (int ci = 0; ci < CI_; ++ci) s += __expf(l[ci] - m);
    const float logZ = m + __logf(s);

    const int M    = co & 15;
    const int tile = kidx * 2 + (co >> 4);
    #pragma unroll
    for (int ci = 0; ci < CI_; ++ci) {
        const float w  = __expf(l[ci] - logZ);           // softmax weight
        const int g    = ci >> 4;                        // lane group holding this K
        const int lane = M + 16 * g;
        const int j    = ci & 15;                        // half index within lane
        wsA[(tile * 32 + lane) * 16 + j] = (_Float16)w;
    }
}

// ---------------------------------------------------------------------------
// Main: one wave32 per {n, h, 32-wide w span}. Each lane owns an adjacent
// even/odd pixel pair (float2 loads/stores -> fully coalesced b64), builds two
// B operands (k2=0 and k2=1 pixel tiles), and runs 4 WMMAs:
//   {k2=0,1} x {co 0-15, co 16-31}.
//   y[n,co,h,w] = m + log( sum_ci W[co,ci,kidx] * exp(x[n,ci,h,w] - m) )
// B operand (32x16 KxN, 16-bit): lane L holds column N=L%16, K = L<16 ? 0..15
// : 16..31 in halfs 0..15. D layout: lane L, vgpr r -> (M=r+8*(L>=16), N=L%16),
// which is the lane's own pixel pair, so m_e/m_o are already resident.
// ---------------------------------------------------------------------------
__global__ void __launch_bounds__(256)
sumconv_main_kernel(const float* __restrict__ x,
                    const _Float16* __restrict__ wsA,
                    float* __restrict__ out) {
    const int wave = (blockIdx.x * 256 + threadIdx.x) >> 5;   // 0..4095
    const int lane = threadIdx.x & 31;

    // wave bits: [0]=w-span, [6:1]=h, [11:7]=n
    const int t  = wave & 1;
    const int h  = (wave >> 1) & 63;
    const int n  = (wave >> 7) & 31;
    const int k1 = h & 1;

    const int pcol   = lane & 15;                  // pixel column N of this lane
    const int w0     = t * 32 + (pcol << 1);       // even pixel; w0+1 = odd pixel
    const int cibase = lane & 16;                  // CI half: 0 or 16

    // Coalesced gather: 16 x b64, dense 128B per half-wave per CI row.
    const float* xp = x + (n * CI_ + cibase) * HW_ + h * W_ + w0;
    float xe[16], xo[16];
    #pragma unroll
    for (int j = 0; j < 16; ++j) {
        const float2 v = *(const float2*)(xp + j * HW_);
        xe[j] = v.x;
        xo[j] = v.y;
    }

    // Per-pixel max over all 32 CI: local 16-way, then lane-pair via shfl_xor.
    float me = xe[0], mo = xo[0];
    #pragma unroll
    for (int j = 1; j < 16; ++j) { me = fmaxf(me, xe[j]); mo = fmaxf(mo, xo[j]); }
    me = fmaxf(me, __shfl_xor(me, 16, 32));
    mo = fmaxf(mo, __shfl_xor(mo, 16, 32));

    // B operands: exp(x - m) in f16, one per k2 parity.
    v16h be, bo;
    #pragma unroll
    for (int j = 0; j < 16; ++j) {
        be[j] = (_Float16)__expf(xe[j] - me);
        bo[j] = (_Float16)__expf(xo[j] - mo);
    }

    // A tiles for kidx = k1*2 + {0,1}, both co-halves. 8 KB total, hot in L2.
    const v16h* A   = (const v16h*)wsA;
    const int  base = (k1 * 2) * 2 * 32;           // tile stride 32 v16h
    const v16h ae0  = A[base + 0 * 32 + lane];     // k2=0, co 0-15
    const v16h ae1  = A[base + 1 * 32 + lane];     // k2=0, co 16-31
    const v16h ao0  = A[base + 2 * 32 + lane];     // k2=1, co 0-15
    const v16h ao1  = A[base + 3 * 32 + lane];     // k2=1, co 16-31

    v8f ce0 = {}, ce1 = {}, co0 = {}, co1 = {};
    ce0 = __builtin_amdgcn_wmma_f32_16x16x32_f16(false, ae0, false, be,
                                                 (short)0, ce0, false, false);
    ce1 = __builtin_amdgcn_wmma_f32_16x16x32_f16(false, ae1, false, be,
                                                 (short)0, ce1, false, false);
    co0 = __builtin_amdgcn_wmma_f32_16x16x32_f16(false, ao0, false, bo,
                                                 (short)0, co0, false, false);
    co1 = __builtin_amdgcn_wmma_f32_16x16x32_f16(false, ao1, false, bo,
                                                 (short)0, co1, false, false);

    // Coalesced scatter: even/odd outputs for a given co are w-adjacent -> b64.
    const int Mofs = (lane >> 4) << 3;             // 0 or 8
    float* op = out + (n * CO_) * HW_ + h * W_ + w0;
    #pragma unroll
    for (int r = 0; r < 8; ++r) {
        const int cA = r + Mofs;                   // co 0-15 half
        const int cB = cA + 16;                    // co 16-31 half
        float2 vA, vB;
        vA.x = me + __logf(ce0[r]);  vA.y = mo + __logf(co0[r]);
        vB.x = me + __logf(ce1[r]);  vB.y = mo + __logf(co1[r]);
        *(float2*)(op + cA * HW_) = vA;
        *(float2*)(op + cB * HW_) = vB;
    }
}

extern "C" void kernel_launch(void* const* d_in, const int* in_sizes, int n_in,
                              void* d_out, int out_size, void* d_ws, size_t ws_size,
                              hipStream_t stream) {
    (void)in_sizes; (void)n_in; (void)out_size; (void)ws_size;
    const float* x      = (const float*)d_in[0];   // [32,32,64,64] f32
    const float* logits = (const float*)d_in[1];   // [32,32,2,2]  f32
    float*       y      = (float*)d_out;           // [32,32,64,64] f32
    _Float16*    wsA    = (_Float16*)d_ws;         // 8 KB of A-layout weights

    sumconv_prep_kernel<<<1, 128, 0, stream>>>(logits, wsA);

    // 4096 wave-tiles (32 n * 64 h * 2 w-spans), 8 waves per block.
    sumconv_main_kernel<<<512, 256, 0, stream>>>(x, wsA, y);
}
